// SigmoidAttentionJoiner_47407849013430
// MI455X (gfx1250) — matrix operand order repaired and checked by reading, not compiled, not run on hardware
//
#include <hip/hip_runtime.h>
#include <math.h>

#define J_DIM 512
#define V_DIM 2000

typedef __attribute__((ext_vector_type(16))) __bf16 v16bf;
typedef __attribute__((ext_vector_type(8)))  float  v8f;

__device__ __forceinline__ unsigned short f2bf(float f) {
    unsigned int u = __float_as_uint(f);
    u += 0x7FFFu + ((u >> 16) & 1u);       // round to nearest even
    return (unsigned short)(u >> 16);
}
__device__ __forceinline__ float bf2f(unsigned short h) {
    return __uint_as_float(((unsigned int)h) << 16);
}

union Frag {
    v16bf v;
    uint4 q[2];
    unsigned short us[16];
};

// ---- A-fragment loaders (16x32 bf16 tile, per-ISA lane layout) -------------
// lane<16 : row=lane,    K = k0..k0+7  and k0+16..k0+23
// lane>=16: row=lane-16, K = k0+8..k0+15 and k0+24..k0+31
__device__ __forceinline__ void load_afrag(const float* A, int K, int r, int k0, Frag& f) {
    const float4* p0 = (const float4*)(A + (size_t)r * K + k0);
    float4 x0 = p0[0], x1 = p0[1];
    const float4* p1 = (const float4*)(A + (size_t)r * K + k0 + 16);
    float4 x2 = p1[0], x3 = p1[1];
    f.us[0]=f2bf(x0.x);  f.us[1]=f2bf(x0.y);  f.us[2]=f2bf(x0.z);  f.us[3]=f2bf(x0.w);
    f.us[4]=f2bf(x1.x);  f.us[5]=f2bf(x1.y);  f.us[6]=f2bf(x1.z);  f.us[7]=f2bf(x1.w);
    f.us[8]=f2bf(x2.x);  f.us[9]=f2bf(x2.y);  f.us[10]=f2bf(x2.z); f.us[11]=f2bf(x2.w);
    f.us[12]=f2bf(x3.x); f.us[13]=f2bf(x3.y); f.us[14]=f2bf(x3.z); f.us[15]=f2bf(x3.w);
}
__device__ __forceinline__ void load_afrag(const unsigned short* A, int K, int r, int k0, Frag& f) {
    f.q[0] = *(const uint4*)(A + (size_t)r * K + k0);
    f.q[1] = *(const uint4*)(A + (size_t)r * K + k0 + 16);
}

// ---- async DMA: one 16B chunk global -> LDS (gfx1250, ASYNCcnt-tracked) ----
__device__ __forceinline__ void async_b128(unsigned int lds_off, const void* gsrc) {
    asm volatile("global_load_async_to_lds_b128 %0, %1, off"
                 :: "v"(lds_off), "v"((unsigned long long)(uintptr_t)gsrc)
                 : "memory");
}
__device__ __forceinline__ void wait_async0() {
    asm volatile("s_wait_asynccnt 0x0" ::: "memory");
}

// Stage one 64-col x 64-k bf16 B tile into LDS (8KB), 2 async b128 per thread.
// LDS layout: [row][k], row stride 64 elements.
__device__ __forceinline__ void stage_btile(const unsigned short* __restrict__ B,
                                            int N, int K, int nb, int kt,
                                            unsigned short* dst /* shared */) {
    const int tid = threadIdx.x;
#pragma unroll
    for (int i = 0; i < 2; ++i) {
        int c   = tid + i * 256;       // chunk 0..511
        int row = c >> 3;              // 0..63
        int kc  = (c & 7) * 8;         // 0..56
        int n   = nb + row;
        n = (n < N) ? n : (N - 1);     // clamp (no EXEC divergence)
        const unsigned short* src = B + (size_t)n * K + kt + kc;
        unsigned int lds = (unsigned int)(uintptr_t)(dst + row * 64 + kc);
        async_b128(lds, src);
    }
}

// ---- GEMM: C[M,N] = A[M,K] * B[N,K]^T + bias; B is bf16 row-major [N,K] ----
// EPI 0: store bf16.  EPI 1: store bf16 tanh(dec + x).  EPI 2: store fp32.
// Block: 256 thr = 8 waves. Wave w owns 2 M-tiles (32 rows) x 4 N-tiles;
// block covers 256 rows x 64 cols. B tile staged once per block via async-LDS,
// double buffered; 8 independent WMMAs per K-step hide LDS/global latency.
// Grid must exactly cover M (no early return: barriers + full-EXEC WMMA).
template <typename AT, int EPI>
__global__ __launch_bounds__(256)
void gemm_kernel(const AT* __restrict__ A,
                 const unsigned short* __restrict__ B,
                 const float* __restrict__ bias,
                 const unsigned short* __restrict__ dec,
                 void* __restrict__ Cout,
                 int M, int N, int K)
{
    __shared__ unsigned short Bsh[2][64 * 64];   // 2 x 8KB

    const int lane  = threadIdx.x & 31;
    const int wave  = threadIdx.x >> 5;
    const int rbase = (blockIdx.y * 8 + wave) * 32;   // 2 M-tiles per wave
    const int nb    = blockIdx.x * 64;

    const int ar = lane & 15;            // A row within tile
    const int ak = (lane >> 4) * 8;      // A k sub-offset
    const int bn = lane & 15;            // B col within tile
    const int bk = (lane >> 4) * 8;      // B k sub-offset

    v8f zero = {0.f,0.f,0.f,0.f,0.f,0.f,0.f,0.f};
    v8f acc[8] = {zero, zero, zero, zero, zero, zero, zero, zero}; // [m*4+t]

    const int nk = K >> 6;               // number of 64-wide K tiles
    stage_btile(B, N, K, nb, 0, &Bsh[0][0]);
    int buf = 0;

    for (int it = 0; it < nk; ++it) {
        wait_async0();                   // my async chunks for tile `it` done
        __syncthreads();                 // => whole tile visible to all waves
        if (it + 1 < nk)                 // uniform branch
            stage_btile(B, N, K, nb, (it + 1) << 6, &Bsh[buf ^ 1][0]);

        const int kt = it << 6;
        // speculative prefetch of next A K-tile (global_prefetch_b8)
        __builtin_prefetch(A + (size_t)(rbase + ar) * K + kt + 64, 0, 1);
        __builtin_prefetch(A + (size_t)(rbase + 16 + ar) * K + kt + 64, 0, 1);

#pragma unroll
        for (int k2 = 0; k2 < 64; k2 += 32) {
            Frag af0, af1;
            load_afrag(A, K, rbase + ar,      ak + kt + k2, af0);
            load_afrag(A, K, rbase + 16 + ar, ak + kt + k2, af1);
            Frag bfm[4];
#pragma unroll
            for (int t = 0; t < 4; ++t) {
                const uint4* bp = (const uint4*)(&Bsh[buf][(t * 16 + bn) * 64 + bk + k2]);
                bfm[t].q[0] = bp[0];     // ds_load_b128
                bfm[t].q[1] = bp[2];
            }
#pragma unroll
            for (int t = 0; t < 4; ++t)
                acc[t] = __builtin_amdgcn_wmma_f32_16x16x32_bf16(
                    false, af0.v, false, bfm[t].v, (short)0, acc[t], false, false);
#pragma unroll
            for (int t = 0; t < 4; ++t)
                acc[4 + t] = __builtin_amdgcn_wmma_f32_16x16x32_bf16(
                    false, af1.v, false, bfm[t].v, (short)0, acc[4 + t], false, false);
        }
        buf ^= 1;
        // no bottom barrier needed: next iteration's (wait + barrier) orders
        // everyone past compute(it) before tile (it+2) overwrites this buffer
    }

    const int crow0 = (lane >> 4) * 8;   // C layout: lanes 0-15 rows 0-7, 16-31 rows 8-15
#pragma unroll
    for (int m = 0; m < 2; ++m) {
#pragma unroll
        for (int t = 0; t < 4; ++t) {
            int col = nb + t * 16 + (lane & 15);
            if (col >= N) continue;
            float bv = bias ? bias[col] : 0.f;
#pragma unroll
            for (int i = 0; i < 8; ++i) {
                int row = rbase + m * 16 + crow0 + i;
                float val = acc[m * 4 + t][i] + bv;
                size_t idx = (size_t)row * N + col;
                if (EPI == 2) {
                    ((float*)Cout)[idx] = val;
                } else if (EPI == 1) {
                    float d = bf2f(dec[idx]);
                    ((unsigned short*)Cout)[idx] = f2bf(tanhf(d + val));
                } else {
                    ((unsigned short*)Cout)[idx] = f2bf(val);
                }
            }
        }
    }
}

// ---- LayerNorm: one wave per 512-elem row (16 elems / lane) ----------------
__global__ __launch_bounds__(256)
void ln_kernel(const unsigned short* __restrict__ x, unsigned short* __restrict__ y,
               const float* __restrict__ g, const float* __restrict__ b, int rows)
{
    int gw   = (int)((blockIdx.x * blockDim.x + threadIdx.x) >> 5);
    int lane = threadIdx.x & 31;
    if (gw >= rows) return;
    const unsigned short* xr = x + (size_t)gw * J_DIM;
    int c0 = lane * 16;
    Frag f;
    f.q[0] = *(const uint4*)(xr + c0);
    f.q[1] = *(const uint4*)(xr + c0 + 8);
    float v[16], s = 0.f, s2 = 0.f;
#pragma unroll
    for (int j = 0; j < 16; ++j) { v[j] = bf2f(f.us[j]); s += v[j]; s2 += v[j] * v[j]; }
    for (int m = 16; m >= 1; m >>= 1) { s += __shfl_xor(s, m, 32); s2 += __shfl_xor(s2, m, 32); }
    float mu  = s  * (1.f / J_DIM);
    float var = s2 * (1.f / J_DIM) - mu * mu;
    float rs  = rsqrtf(var + 1e-5f);
    Frag o;
#pragma unroll
    for (int j = 0; j < 16; ++j)
        o.us[j] = f2bf((v[j] - mu) * rs * g[c0 + j] + b[c0 + j]);
    unsigned short* yr = y + (size_t)gw * J_DIM;
    *(uint4*)(yr + c0)     = o.q[0];
    *(uint4*)(yr + c0 + 8) = o.q[1];
}

// ---- Sigmoid attention: wave per row; head = lane/4 (64 dims = 4 lanes) ----
__global__ __launch_bounds__(256)
void attn_kernel(const unsigned short* __restrict__ q, const unsigned short* __restrict__ k,
                 const unsigned short* __restrict__ v, unsigned short* __restrict__ ctx, int rows)
{
    int gw   = (int)((blockIdx.x * blockDim.x + threadIdx.x) >> 5);
    int lane = threadIdx.x & 31;
    if (gw >= rows) return;
    size_t base = (size_t)gw * J_DIM + (size_t)lane * 16;
    Frag fq, fk, fv;
    fq.q[0] = *(const uint4*)(q + base); fq.q[1] = *(const uint4*)(q + base + 8);
    fk.q[0] = *(const uint4*)(k + base); fk.q[1] = *(const uint4*)(k + base + 8);
    fv.q[0] = *(const uint4*)(v + base); fv.q[1] = *(const uint4*)(v + base + 8);
    float d = 0.f, vf[16];
#pragma unroll
    for (int j = 0; j < 16; ++j) {
        vf[j] = bf2f(fv.us[j]);
        d += bf2f(fq.us[j]) * bf2f(fk.us[j]);
    }
    d += __shfl_xor(d, 1, 32);   // reduce within the 4-lane head group
    d += __shfl_xor(d, 2, 32);
    float a = 1.f / (1.f + __expf(-d * 0.125f));   // scale = 64^-0.5
    Frag o;
#pragma unroll
    for (int j = 0; j < 16; ++j) o.us[j] = f2bf(a * vf[j]);
    *(uint4*)(ctx + base)     = o.q[0];
    *(uint4*)(ctx + base + 8) = o.q[1];
}

// ---- fp32 -> bf16 weight conversion ----------------------------------------
__global__ void cvt_kernel(const float* __restrict__ s, unsigned short* __restrict__ d, int n) {
    for (int i = blockIdx.x * blockDim.x + threadIdx.x; i < n; i += gridDim.x * blockDim.x)
        d[i] = f2bf(s[i]);
}

extern "C" void kernel_launch(void* const* d_in, const int* in_sizes, int n_in,
                              void* d_out, int out_size, void* d_ws, size_t ws_size,
                              hipStream_t stream) {
    const float* enc_in = (const float*)d_in[0];
    const float* dec_in = (const float*)d_in[1];
    const float* We     = (const float*)d_in[2];
    const float* be     = (const float*)d_in[3];
    const float* Wd     = (const float*)d_in[4];
    const float* bd     = (const float*)d_in[5];
    const float* g_q    = (const float*)d_in[6];
    const float* beta_q = (const float*)d_in[7];
    const float* g_kv   = (const float*)d_in[8];
    const float* beta_kv= (const float*)d_in[9];
    const float* Wq     = (const float*)d_in[10];
    const float* bq     = (const float*)d_in[11];
    const float* Wk     = (const float*)d_in[12];
    const float* bk     = (const float*)d_in[13];
    const float* Wv     = (const float*)d_in[14];
    const float* bv     = (const float*)d_in[15];
    const float* Wo     = (const float*)d_in[16];
    const float* bo     = (const float*)d_in[17];
    const float* Wout   = (const float*)d_in[18];
    const float* bout   = (const float*)d_in[19];

    const int ROWS = in_sizes[0] / J_DIM;   // 40960

    // ---- workspace layout (256B aligned) ----
    char* ws = (char*)d_ws;
    size_t off = 0;
    auto take = [&](size_t bytes) {
        char* p = ws + off;
        off = (off + bytes + 255) & ~(size_t)255;
        return p;
    };
    const size_t WSQ   = (size_t)J_DIM * J_DIM * 2;      // 512KB bf16 weight
    const size_t ACT   = (size_t)ROWS * J_DIM * 2;       // 40MB bf16 activation
    unsigned short* Web   = (unsigned short*)take(WSQ);
    unsigned short* Wdb   = (unsigned short*)take(WSQ);
    unsigned short* Wqb   = (unsigned short*)take(WSQ);
    unsigned short* Wkb   = (unsigned short*)take(WSQ);
    unsigned short* Wvb   = (unsigned short*)take(WSQ);
    unsigned short* Wob   = (unsigned short*)take(WSQ);
    unsigned short* Woutb = (unsigned short*)take((size_t)V_DIM * J_DIM * 2);
    unsigned short* enc_b = (unsigned short*)take(ACT);  // enc -> kv_ln (in place)
    unsigned short* dec_b = (unsigned short*)take(ACT);  // kept for tanh(dec+ctx)
    unsigned short* qln_b = (unsigned short*)take(ACT);  // q_ln -> later h
    unsigned short* q_b   = (unsigned short*)take(ACT);
    unsigned short* k_b   = (unsigned short*)take(ACT);  // k -> later ctx (in place)
    unsigned short* v_b   = (unsigned short*)take(ACT);

    dim3 blk(256);

    // 1) weights fp32 -> bf16
    cvt_kernel<<<256, blk, 0, stream>>>(We,   Web,   J_DIM * J_DIM);
    cvt_kernel<<<256, blk, 0, stream>>>(Wd,   Wdb,   J_DIM * J_DIM);
    cvt_kernel<<<256, blk, 0, stream>>>(Wq,   Wqb,   J_DIM * J_DIM);
    cvt_kernel<<<256, blk, 0, stream>>>(Wk,   Wkb,   J_DIM * J_DIM);
    cvt_kernel<<<256, blk, 0, stream>>>(Wv,   Wvb,   J_DIM * J_DIM);
    cvt_kernel<<<256, blk, 0, stream>>>(Wo,   Wob,   J_DIM * J_DIM);
    cvt_kernel<<<256, blk, 0, stream>>>(Wout, Woutb, V_DIM * J_DIM);

    dim3 gJ(J_DIM / 64, ROWS / 256);                 // (8, 160)
    // 2) input projections (fp32 A, converted in-register)
    gemm_kernel<float, 0><<<gJ, blk, 0, stream>>>(enc_in, Web, be, nullptr, enc_b, ROWS, J_DIM, J_DIM);
    gemm_kernel<float, 0><<<gJ, blk, 0, stream>>>(dec_in, Wdb, bd, nullptr, dec_b, ROWS, J_DIM, J_DIM);

    // 3) LayerNorms (kv in-place; q to its own buffer)
    int lnBlocks = ROWS / 8;                         // 8 waves/block, wave per row
    ln_kernel<<<lnBlocks, blk, 0, stream>>>(enc_b, enc_b, g_kv, beta_kv, ROWS);
    ln_kernel<<<lnBlocks, blk, 0, stream>>>(dec_b, qln_b, g_q,  beta_q,  ROWS);

    // 4) Q/K/V projections
    gemm_kernel<unsigned short, 0><<<gJ, blk, 0, stream>>>(qln_b, Wqb, bq, nullptr, q_b, ROWS, J_DIM, J_DIM);
    gemm_kernel<unsigned short, 0><<<gJ, blk, 0, stream>>>(enc_b, Wkb, bk, nullptr, k_b, ROWS, J_DIM, J_DIM);
    gemm_kernel<unsigned short, 0><<<gJ, blk, 0, stream>>>(enc_b, Wvb, bv, nullptr, v_b, ROWS, J_DIM, J_DIM);

    // 5) per-head sigmoid attention; ctx overwrites k_b
    attn_kernel<<<lnBlocks, blk, 0, stream>>>(q_b, k_b, v_b, k_b, ROWS);

    // 6) output projection + fused h = tanh(dec + ctx@Wo^T + bo); h overwrites qln_b
    gemm_kernel<unsigned short, 1><<<gJ, blk, 0, stream>>>(k_b, Wob, bo, dec_b, qln_b, ROWS, J_DIM, J_DIM);

    // 7) vocab projection -> fp32 logits
    dim3 gV((V_DIM + 63) / 64, ROWS / 256);          // (32, 160), N edge guarded
    gemm_kernel<unsigned short, 2><<<gV, blk, 0, stream>>>(qln_b, Woutb, bout, nullptr, d_out, ROWS, V_DIM, J_DIM);
}
